// NLBlockND_17583596110102
// MI455X (gfx1250) — matrix operand order repaired and checked by reading, not compiled, not run on hardware
//
#include <hip/hip_runtime.h>
#include <math.h>

#define B_   8
#define C_   1024
#define CI_  512
#define N_   3136
#define BN_EPS 1e-5f

typedef __attribute__((ext_vector_type(16))) _Float16 v16h;
typedef __attribute__((ext_vector_type(8)))  _Float16 v8h;
typedef __attribute__((ext_vector_type(8)))  float    v8f;

static __device__ __forceinline__ v16h load16h(const _Float16* lo, const _Float16* hi) {
  union { v16h v; v8h h[2]; } u;
  u.h[0] = *(const v8h*)lo;
  u.h[1] = *(const v8h*)hi;
  return u.v;
}

static __device__ __forceinline__ v8f wmma16(v16h a, v16h b, v8f c) {
  // D = A(16x32 f16) * B(32x16 f16) + C(16x16 f32)
  return __builtin_amdgcn_wmma_f32_16x16x32_f16(false, a, false, b, (short)0, c,
                                                false, false);
}

// ---------------------------------------------------------------------------
// Kernel 0: one-shot fp32 -> f16 conversion (weights; 2.1M elements total).
// Removes all per-tile v_cvt ladders from the GEMM inner loops.
// ---------------------------------------------------------------------------
__global__ __launch_bounds__(256)
void cvt_f32_to_f16(const float* __restrict__ src, _Float16* __restrict__ dst,
                    int n)
{
  const int stride = gridDim.x * 256;
  for (int i = blockIdx.x * 256 + threadIdx.x; i < n; i += stride)
    dst[i] = (_Float16)src[i];
}

// ---------------------------------------------------------------------------
// Kernel 1: projection GEMM.  X is (C_in, N) per batch (channel-major input),
// Wh is (CIo, C_in) f16 (pre-converted), out O is (N, CIo) f16 row-major.
// Workgroup tile 64(M=n) x 64(N=ci); 8 waves = 4 (M) x 2 (32-wide N) blocks.
// X^T tile staged+converted through LDS (row stride 40 halves = 80B, 16B
// aligned, bank-conflict free); weight B-fragments are two b128 global loads.
// ---------------------------------------------------------------------------
__global__ __launch_bounds__(256)
void proj_gemm(const float* __restrict__ X, const _Float16* __restrict__ Wh,
               const float* __restrict__ bias, _Float16* __restrict__ O,
               int Cin, int CIo)
{
  __shared__ _Float16 a_lds[64 * 40];

  const int b   = blockIdx.z;
  const int n0  = blockIdx.x * 64;
  const int ci0 = blockIdx.y * 64;
  const float* Xb = X + (size_t)b * Cin * N_;
  _Float16*    Ob = O + (size_t)b * N_ * CIo;

  const int tid  = threadIdx.x;
  const int lane = tid & 31;
  const int w    = tid >> 5;
  const int wm   = w & 3;   // 16-row block
  const int wn   = w >> 2;  // 32-col block

  const int mloc = wm * 16 + (lane & 15);
  const int kbA  = (lane >> 4) * 16;  // A frag: 16 contiguous K
  const int jcol = lane & 15;
  const int kloB = (lane >> 4) * 16;  // B frag: 16 contiguous K

  v8f acc0 = {}, acc1 = {};

  for (int k0 = 0; k0 < Cin; k0 += 32) {
    __syncthreads();
    // stage X^T (64 n x 32 c) -> LDS f16; reads coalesced along n
    {
      const int n_l = tid & 63;
      const int c_b = (tid >> 6) * 8;
#pragma unroll
      for (int i = 0; i < 8; ++i) {
        float v = Xb[(size_t)(k0 + c_b + i) * N_ + (n0 + n_l)];
        a_lds[n_l * 40 + c_b + i] = (_Float16)v;
      }
      if (k0 + 32 < Cin)  // hide the strided-load latency of the next tile
        __builtin_prefetch(&Xb[(size_t)(k0 + 32 + c_b) * N_ + (n0 + n_l)], 0, 1);
    }
    __syncthreads();

    v16h a = load16h(&a_lds[mloc * 40 + kbA], &a_lds[mloc * 40 + kbA + 8]);
#pragma unroll
    for (int t = 0; t < 2; ++t) {
      const int ci = ci0 + wn * 32 + t * 16 + jcol;
      const _Float16* wp = Wh + (size_t)ci * Cin + k0 + kloB;
      v16h bf = load16h(wp, wp + 8);
      if (t == 0) acc0 = wmma16(a, bf, acc0);
      else        acc1 = wmma16(a, bf, acc1);
    }
  }

  const int row = (lane >> 4) * 8;
  const int col = lane & 15;
#pragma unroll
  for (int t = 0; t < 2; ++t) {
    v8f acc = t ? acc1 : acc0;
    const int ci = ci0 + wn * 32 + t * 16 + col;
    const float bv = bias[ci];
#pragma unroll
    for (int r = 0; r < 8; ++r) {
      const int n = n0 + wm * 16 + row + r;
      Ob[(size_t)n * CIo + ci] = (_Float16)(acc[r] + bv);
    }
  }
}

// ---------------------------------------------------------------------------
// Kernel 2: fused flash attention.  Per block: 32 query rows; stream over all
// 3136 keys in 32-wide column tiles.  S = th*ph^T via WMMA (K split 4-ways
// across waves, combined with ds_add_f32 atomics), online softmax in LDS,
// y += P*G with WMMA accumulators resident in VGPRs (wave owns 16 x 128).
// The f matrix (315 MB at fp32) never touches HBM.
// ---------------------------------------------------------------------------
__global__ __launch_bounds__(256)
void nl_attention(const _Float16* __restrict__ TH, const _Float16* __restrict__ PH,
                  const _Float16* __restrict__ GX, _Float16* __restrict__ Y)
{
  __shared__ _Float16 gT[CI_ * 40];   // transposed G tile: [ci][k], 40 KB
  __shared__ float    S[32 * 32];     // score tile
  __shared__ _Float16 P[32 * 40];     // softmax probs (f16, padded stride)
  __shared__ float    scaleL[32], rmaxL[32], rsumL[32];

  const int b  = blockIdx.y;
  const int n0 = blockIdx.x * 32;
  const _Float16* THb = TH + (size_t)b * N_ * CI_;
  const _Float16* PHb = PH + (size_t)b * N_ * CI_;
  const _Float16* GXb = GX + (size_t)b * N_ * CI_;
  _Float16*       Yb  = Y  + (size_t)b * N_ * CI_;

  const int tid  = threadIdx.x;
  const int lane = tid & 31;
  const int w    = tid >> 5;
  const int wm   = w & 1;    // 16-row block of the 32 query rows
  const int wk   = w >> 1;   // S: K chunk (128);  P@G: 128-wide ci block

  const int kbA  = (lane >> 4) * 16;
  const int jcol = lane & 15;
  const int kloB = (lane >> 4) * 16;
  const int mloc = wm * 16 + (lane & 15);
  const _Float16* Arow = THb + (size_t)(n0 + mloc) * CI_;

  v8f yacc[8] = {};

  if (tid < 32) { rmaxL[tid] = -3.0e38f; rsumL[tid] = 0.0f; }

  for (int c0 = 0; c0 < N_; c0 += 32) {
    __syncthreads();  // prior-tile readers of gT / S are done
    for (int i = tid; i < 32 * 32; i += 256) S[i] = 0.0f;
    // stage G tile (32 keys x 512 ci) transposed into LDS
    for (int idx = tid; idx < 32 * 64; idx += 256) {
      const int r  = idx >> 6;          // key row 0..31
      const int c8 = (idx & 63) * 8;    // ci
      v8h g = *(const v8h*)&GXb[(size_t)(c0 + r) * CI_ + c8];
#pragma unroll
      for (int i = 0; i < 8; ++i) gT[(c8 + i) * 40 + r] = g[i];
      if (c0 + 32 < N_)                 // prefetch next key tile
        __builtin_prefetch(&GXb[(size_t)(c0 + 32 + r) * CI_ + c8], 0, 1);
    }

    // ---- S partial: wave computes 16x32 over its 128-wide K chunk
    v8f s0 = {}, s1 = {};
#pragma unroll
    for (int ks = 0; ks < 4; ++ks) {
      const int k = wk * 128 + ks * 32;
      v16h a = load16h(&Arow[k + kbA], &Arow[k + kbA + 8]);
#pragma unroll
      for (int nt = 0; nt < 2; ++nt) {
        const _Float16* pp = &PHb[(size_t)(c0 + nt * 16 + jcol) * CI_ + k + kloB];
        v16h bf = load16h(pp, pp + 8);
        if (nt == 0) s0 = wmma16(a, bf, s0);
        else         s1 = wmma16(a, bf, s1);
      }
    }
    __syncthreads();  // S zeroed + gT staged everywhere
    {
      const int rb = wm * 16 + (lane >> 4) * 8;
#pragma unroll
      for (int r = 0; r < 8; ++r) {
        atomicAdd(&S[(rb + r) * 32 + jcol],      s0[r]);
        atomicAdd(&S[(rb + r) * 32 + 16 + jcol], s1[r]);
      }
    }
    __syncthreads();

    // ---- online softmax: one thread per query row
    if (tid < 32) {
      const float* Sr = &S[tid * 32];
      float mxo = rmaxL[tid];
      float mx  = mxo;
      for (int j = 0; j < 32; ++j) mx = fmaxf(mx, Sr[j]);
      const float corr = __expf(mxo - mx);   // 0 on first tile (mxo = -3e38)
      float s = rsumL[tid] * corr;
      for (int j = 0; j < 32; ++j) {
        const float p = __expf(Sr[j] - mx);
        s += p;
        P[tid * 40 + j] = (_Float16)p;
      }
      rmaxL[tid] = mx; rsumL[tid] = s; scaleL[tid] = corr;
    }
    __syncthreads();

    // ---- rescale running accumulators, then y += P @ G
    {
      const int rb = wm * 16 + (lane >> 4) * 8;
      float cs[8];
#pragma unroll
      for (int r = 0; r < 8; ++r) cs[r] = scaleL[rb + r];
#pragma unroll
      for (int t = 0; t < 8; ++t)
#pragma unroll
        for (int r = 0; r < 8; ++r) yacc[t][r] *= cs[r];

      v16h a = load16h(&P[mloc * 40 + kbA], &P[mloc * 40 + kbA + 8]);
#pragma unroll
      for (int t = 0; t < 8; ++t) {
        const int ci = wk * 128 + t * 16 + jcol;
        v16h bf = load16h(&gT[ci * 40 + kloB], &gT[ci * 40 + kloB + 8]);
        yacc[t] = wmma16(a, bf, yacc[t]);
      }
    }
  }

  __syncthreads();
  {
    const int rb = wm * 16 + (lane >> 4) * 8;
    float inv[8];
#pragma unroll
    for (int r = 0; r < 8; ++r) inv[r] = 1.0f / rsumL[rb + r];
#pragma unroll
    for (int t = 0; t < 8; ++t) {
      const int ci = wk * 128 + t * 16 + jcol;
#pragma unroll
      for (int r = 0; r < 8; ++r)
        Yb[(size_t)(n0 + rb + r) * CI_ + ci] = (_Float16)(yacc[t][r] * inv[r]);
    }
  }
}

// ---------------------------------------------------------------------------
// Kernel 3: wy = y @ wz^T + wz_b, fp32 output (needed at f32 for BN stats).
// A-fragments come straight from global (y is row-major f16, per-lane 16B
// aligned runs); weight B-fragments are b128 loads of pre-converted f16.
// ---------------------------------------------------------------------------
__global__ __launch_bounds__(256)
void wz_gemm(const _Float16* __restrict__ Yin, const _Float16* __restrict__ Wh,
             const float* __restrict__ bias, float* __restrict__ WY)
{
  const int b  = blockIdx.z;
  const int n0 = blockIdx.x * 64;
  const int c0 = blockIdx.y * 64;
  const int tid = threadIdx.x, lane = tid & 31, w = tid >> 5;
  const int wm = w & 3, wn = w >> 2;
  const int kbA = (lane >> 4) * 16, jcol = lane & 15, kloB = (lane >> 4) * 16;

  const _Float16* Ybase =
      Yin + ((size_t)b * N_ + n0 + wm * 16 + (lane & 15)) * CI_;

  v8f acc0 = {}, acc1 = {};
  for (int k0 = 0; k0 < CI_; k0 += 32) {
    v16h a = load16h(&Ybase[k0 + kbA], &Ybase[k0 + kbA + 8]);
#pragma unroll
    for (int t = 0; t < 2; ++t) {
      const int c = c0 + wn * 32 + t * 16 + jcol;
      const _Float16* wp = Wh + (size_t)c * CI_ + k0 + kloB;
      v16h bf = load16h(wp, wp + 8);
      if (t == 0) acc0 = wmma16(a, bf, acc0);
      else        acc1 = wmma16(a, bf, acc1);
    }
  }

  const int row = (lane >> 4) * 8, col = lane & 15;
#pragma unroll
  for (int t = 0; t < 2; ++t) {
    v8f acc = t ? acc1 : acc0;
    const int c = c0 + wn * 32 + t * 16 + col;
    const float bv = bias[c];
#pragma unroll
    for (int r = 0; r < 8; ++r) {
      const size_t n = (size_t)b * N_ + n0 + wm * 16 + row + r;
      WY[n * C_ + c] = acc[r] + bv;
    }
  }
}

// ---------------------------------------------------------------------------
// Kernel 4: per-channel batch-norm statistics (sum, sum of squares).
// grid (C_/256, 32 row-chunks); coalesced along channels; global atomics.
// ---------------------------------------------------------------------------
__global__ __launch_bounds__(256)
void bn_stats(const float* __restrict__ WY, float* __restrict__ sums,
              float* __restrict__ sumsq)
{
  const int c = blockIdx.x * 256 + threadIdx.x;
  const int rowsPer = (B_ * N_) / 32;     // 784
  const int r0 = blockIdx.y * rowsPer;
  float s = 0.0f, q = 0.0f;
  for (int r = r0; r < r0 + rowsPer; ++r) {
    const float v = WY[(size_t)r * C_ + c];
    s += v; q += v * v;
  }
  atomicAdd(&sums[c], s);
  atomicAdd(&sumsq[c], q);
}

// ---------------------------------------------------------------------------
// Kernel 5: normalize + residual + transpose back to NCHW.  64x64 LDS tile
// (stride 65 -> bank-conflict-free); both global sides coalesced.
// ---------------------------------------------------------------------------
__global__ __launch_bounds__(256)
void bn_finalize(const float* __restrict__ WY, const float* __restrict__ sums,
                 const float* __restrict__ sumsq, const float* __restrict__ gamma,
                 const float* __restrict__ beta, const float* __restrict__ X1,
                 float* __restrict__ OUT)
{
  __shared__ float tile[64 * 65];
  const int b = blockIdx.z, n0 = blockIdx.x * 64, c0 = blockIdx.y * 64;
  const int tid = threadIdx.x;

  {
    const int cl = tid & 63;
    const int nb = (tid >> 6) * 16;
    const int c = c0 + cl;
    const float inv = 1.0f / (float)(B_ * N_);
    const float mu  = sums[c] * inv;
    const float var = sumsq[c] * inv - mu * mu;
    const float a   = rsqrtf(var + BN_EPS) * gamma[c];
    const float bb  = beta[c] - mu * a;
#pragma unroll
    for (int i = 0; i < 16; ++i) {
      const int nl = nb + i;
      const float v = WY[((size_t)b * N_ + n0 + nl) * C_ + c];
      tile[nl * 65 + cl] = v * a + bb;
    }
  }
  __syncthreads();
  {
    const int nl = tid & 63;
    const int cb = (tid >> 6) * 16;
#pragma unroll
    for (int i = 0; i < 16; ++i) {
      const int cl = cb + i;
      const size_t o = (size_t)b * C_ * N_ + (size_t)(c0 + cl) * N_ + n0 + nl;
      OUT[o] = tile[nl * 65 + cl] + X1[o];
    }
  }
}

// ---------------------------------------------------------------------------
extern "C" void kernel_launch(void* const* d_in, const int* in_sizes, int n_in,
                              void* d_out, int out_size, void* d_ws, size_t ws_size,
                              hipStream_t stream) {
  (void)in_sizes; (void)n_in; (void)out_size; (void)ws_size;
  const float* x1    = (const float*)d_in[0];
  const float* x2    = (const float*)d_in[1];
  const float* g_w   = (const float*)d_in[2];
  const float* g_b   = (const float*)d_in[3];
  const float* th_w  = (const float*)d_in[4];
  const float* th_b  = (const float*)d_in[5];
  const float* ph_w  = (const float*)d_in[6];
  const float* ph_b  = (const float*)d_in[7];
  const float* wz_w  = (const float*)d_in[8];
  const float* wz_b  = (const float*)d_in[9];
  const float* gamma = (const float*)d_in[10];
  const float* beta  = (const float*)d_in[11];
  float* out = (float*)d_out;

  char* ws = (char*)d_ws;
  const size_t SZ   = (size_t)B_ * N_ * CI_ * sizeof(_Float16);  // 25.7 MB
  const size_t WYSZ = (size_t)B_ * N_ * C_ * sizeof(float);      // 102.8 MB
  const size_t WSZ  = (size_t)CI_ * C_ * sizeof(_Float16);       // 1 MB
  _Float16* th = (_Float16*)(ws);
  _Float16* ph = (_Float16*)(ws + SZ);
  _Float16* gx = (_Float16*)(ws + 2 * SZ);
  _Float16* y  = (_Float16*)(ws + 3 * SZ);
  float* wy    = (float*)(ws + 4 * SZ);
  float* sums  = (float*)(ws + 4 * SZ + WYSZ);
  float* sumsq = sums + C_;
  _Float16* g_wh  = (_Float16*)(ws + 4 * SZ + WYSZ + 4096);
  _Float16* th_wh = (_Float16*)((char*)g_wh  + WSZ);
  _Float16* ph_wh = (_Float16*)((char*)th_wh + WSZ);
  _Float16* wz_wh = (_Float16*)((char*)ph_wh + WSZ);

  // one-shot weight conversion (4.2 MB of f16 total)
  const int nw = CI_ * C_;
  cvt_f32_to_f16<<<512, 256, 0, stream>>>(g_w,  g_wh,  nw);
  cvt_f32_to_f16<<<512, 256, 0, stream>>>(th_w, th_wh, nw);
  cvt_f32_to_f16<<<512, 256, 0, stream>>>(ph_w, ph_wh, nw);
  cvt_f32_to_f16<<<512, 256, 0, stream>>>(wz_w, wz_wh, nw);

  const dim3 gp(N_ / 64, CI_ / 64, B_);   // 49 x 8 x 8
  proj_gemm<<<gp, 256, 0, stream>>>(x2, g_wh,  g_b,  gx, C_, CI_);
  proj_gemm<<<gp, 256, 0, stream>>>(x1, th_wh, th_b, th, C_, CI_);
  proj_gemm<<<gp, 256, 0, stream>>>(x2, ph_wh, ph_b, ph, C_, CI_);

  nl_attention<<<dim3(N_ / 32, B_), 256, 0, stream>>>(th, ph, gx, y);

  wz_gemm<<<dim3(N_ / 64, C_ / 64, B_), 256, 0, stream>>>(y, wz_wh, wz_b, wy);

  hipMemsetAsync(sums, 0, 2 * C_ * sizeof(float), stream);
  bn_stats<<<dim3(C_ / 256, 32), 256, 0, stream>>>(wy, sums, sumsq);

  bn_finalize<<<dim3(N_ / 64, C_ / 64, B_), 256, 0, stream>>>(
      wy, sums, sumsq, gamma, beta, x1, out);
}
